// CNN_GNN_Hybrid_21758304321731
// MI455X (gfx1250) — matrix operand order repaired and checked by reading, not compiled, not run on hardware
//
#include <hip/hip_runtime.h>
#include <math.h>

typedef __attribute__((ext_vector_type(16))) _Float16 v16h;
typedef __attribute__((ext_vector_type(8)))  _Float16 v8h;
typedef __attribute__((ext_vector_type(8)))  float    v8f;

__device__ __forceinline__ float sigf(float x) { return 1.0f / (1.0f + expf(-x)); }

__device__ __forceinline__ v16h pack16(v8h lo, v8h hi) {
  v16h r;
#pragma unroll
  for (int i = 0; i < 8; ++i) { r[i] = lo[i]; r[i + 8] = hi[i]; }
  return r;
}

__device__ __forceinline__ v8f wmma_f16(v16h a, v16h b, v8f c) {
  return __builtin_amdgcn_wmma_f32_16x16x32_f16(false, a, false, b, (short)0, c, false, false);
}

// ---------------------------------------------------------------------------
// Problem constants
// ---------------------------------------------------------------------------
#define Bv   4
#define Tv   24
#define Cv   8
#define Hv   57
#define Wv   69
#define Nv   20000
#define Ev   640000
#define CH1  32
#define CH2  64
#define P1H  28   // 57/2
#define P1W  34   // 69/2
#define P2H  14
#define P2W  17
#define FLAT 15232  // 64*14*17
#define CLH  256
#define GH   64
#define ZLH  128
#define BT   (Bv*Tv)          // 96
#define CONVA_ELEMS (BT*CH1*P1H*P1W)   // 2,924,544
#define F_ELEMS     (BT*FLAT)          // 1,462,272 ... wait: 96*15232 = 1,462,272

// ---------------------------------------------------------------------------
// conv1 + relu + maxpool2 : era5[96,8,57,69] -> convA[96,32,28,34]
// ---------------------------------------------------------------------------
__global__ __launch_bounds__(256) void conv1_pool(const float* __restrict__ x,
                                                  const float* __restrict__ w,
                                                  const float* __restrict__ bias,
                                                  float* __restrict__ y) {
  int idx = blockIdx.x * 256 + threadIdx.x;
  if (idx >= CONVA_ELEMS) return;
  int pw  = idx % P1W;  int t1 = idx / P1W;
  int ph  = t1 % P1H;   int t2 = t1 / P1H;
  int oc  = t2 % CH1;   int img = t2 / CH1;
  const float* xb = x + (size_t)img * Cv * Hv * Wv;
  const float* wb = w + oc * Cv * 9;
  float b = bias[oc];
  float best = 0.0f;
#pragma unroll
  for (int dy = 0; dy < 2; ++dy)
#pragma unroll
  for (int dx = 0; dx < 2; ++dx) {
    int h0 = 2 * ph + dy, w0 = 2 * pw + dx;
    float s = b;
    for (int ic = 0; ic < Cv; ++ic) {
      const float* xc = xb + ic * Hv * Wv;
      const float* wc = wb + ic * 9;
#pragma unroll
      for (int kh = 0; kh < 3; ++kh) {
        int ih = h0 + kh - 1;
        if (ih < 0 || ih >= Hv) continue;
#pragma unroll
        for (int kw = 0; kw < 3; ++kw) {
          int iw = w0 + kw - 1;
          if (iw < 0 || iw >= Wv) continue;
          s += xc[ih * Wv + iw] * wc[kh * 3 + kw];
        }
      }
    }
    best = fmaxf(best, fmaxf(s, 0.0f));
  }
  y[idx] = best;
}

// ---------------------------------------------------------------------------
// conv2 + relu + maxpool2 : convA[96,32,28,34] -> fA(f16)[96,15232]
// ---------------------------------------------------------------------------
__global__ __launch_bounds__(256) void conv2_pool(const float* __restrict__ x,
                                                  const float* __restrict__ w,
                                                  const float* __restrict__ bias,
                                                  _Float16* __restrict__ y) {
  int idx = blockIdx.x * 256 + threadIdx.x;
  if (idx >= BT * CH2 * P2H * P2W) return;
  int pw  = idx % P2W;  int t1 = idx / P2W;
  int ph  = t1 % P2H;   int t2 = t1 / P2H;
  int oc  = t2 % CH2;   int img = t2 / CH2;
  const float* xb = x + (size_t)img * CH1 * P1H * P1W;
  const float* wb = w + oc * CH1 * 9;
  float b = bias[oc];
  float best = 0.0f;
#pragma unroll
  for (int dy = 0; dy < 2; ++dy)
#pragma unroll
  for (int dx = 0; dx < 2; ++dx) {
    int h0 = 2 * ph + dy, w0 = 2 * pw + dx;
    float s = b;
    for (int ic = 0; ic < CH1; ++ic) {
      const float* xc = xb + ic * P1H * P1W;
      const float* wc = wb + ic * 9;
#pragma unroll
      for (int kh = 0; kh < 3; ++kh) {
        int ih = h0 + kh - 1;
        if (ih < 0 || ih >= P1H) continue;
#pragma unroll
        for (int kw = 0; kw < 3; ++kw) {
          int iw = w0 + kw - 1;
          if (iw < 0 || iw >= P1W) continue;
          s += xc[ih * P1W + iw] * wc[kh * 3 + kw];
        }
      }
    }
    best = fmaxf(best, fmaxf(s, 0.0f));
  }
  y[idx] = (_Float16)best;  // idx == img*FLAT + oc*238 + ph*17 + pw (NCHW flatten)
}

// ---------------------------------------------------------------------------
// xW = fA[96,15232] @ clstm_wih.T + (bih+bhh)  -> [96,1024]  (WMMA f16)
// 48 blocks x 256 thr: wave = one 16x16 tile, K loop 476 steps of 32.
// ---------------------------------------------------------------------------
__global__ __launch_bounds__(256) void xw_wmma(const _Float16* __restrict__ fA,
                                               const float* __restrict__ wih,
                                               const float* __restrict__ bih,
                                               const float* __restrict__ bhh,
                                               float* __restrict__ xW) {
  int wave = threadIdx.x >> 5, lane = threadIdx.x & 31;
  int g = lane >> 4, col = lane & 15;
  int mt = blockIdx.x >> 3;                  // 0..5
  int nt = (blockIdx.x & 7) * 8 + wave;      // 0..63
  const _Float16* arow = fA + (size_t)(mt * 16 + col) * FLAT;  // A row m=col
  const float*    brow = wih + (size_t)(nt * 16 + col) * FLAT; // B col n=col (B[k][n]=wih[n][k])
  v8f acc = {};
#pragma unroll 2
  for (int k0 = 0; k0 < FLAT; k0 += 32) {
    v8h alo = *reinterpret_cast<const v8h*>(arow + k0 + 8 * g);
    v8h ahi = *reinterpret_cast<const v8h*>(arow + k0 + 16 + 8 * g);
    v16h A = pack16(alo, ahi);
    const float* bp = brow + k0 + 8 * g;     // K(j,g) = 8g + j (+8 for j>=8)
    v16h Bm;
#pragma unroll
    for (int i = 0; i < 16; ++i) Bm[i] = (_Float16)bp[i + (i >= 8 ? 8 : 0)];
    acc = wmma_f16(A, Bm, acc);
  }
  int n = nt * 16 + col;
  float biasv = bih[n] + bhh[n];
#pragma unroll
  for (int r = 0; r < 8; ++r) {
    int m = mt * 16 + r + 8 * g;             // C/D layout: M = r + 8*(lane>=16)
    xW[m * 1024 + n] = acc[r] + biasv;
  }
}

// ---------------------------------------------------------------------------
// clstm recurrence (B=4, hidden 256) + era5 base = fc_b + era5_h . fc_w[:256]
// single block, 1024 threads
// ---------------------------------------------------------------------------
__global__ __launch_bounds__(1024) void clstm_kernel(const float* __restrict__ xW,
                                                     const float* __restrict__ whh,
                                                     const float* __restrict__ fcw,
                                                     const float* __restrict__ fcb,
                                                     float* __restrict__ base) {
  __shared__ float h[4 * CLH];
  __shared__ float c[4 * CLH];
  __shared__ float gate[4 * 1024];
  int tid = threadIdx.x;
  h[tid] = 0.0f; c[tid] = 0.0f;
  __syncthreads();
  for (int t = 0; t < Tv; ++t) {
    float a0 = xW[(0 * Tv + t) * 1024 + tid];
    float a1 = xW[(1 * Tv + t) * 1024 + tid];
    float a2 = xW[(2 * Tv + t) * 1024 + tid];
    float a3 = xW[(3 * Tv + t) * 1024 + tid];
    const float* wr = whh + tid * CLH;
    for (int k = 0; k < CLH; ++k) {
      float wv = wr[k];
      a0 += h[k] * wv; a1 += h[CLH + k] * wv;
      a2 += h[2 * CLH + k] * wv; a3 += h[3 * CLH + k] * wv;
    }
    gate[tid] = a0; gate[1024 + tid] = a1; gate[2048 + tid] = a2; gate[3072 + tid] = a3;
    __syncthreads();
    int b = tid >> 8, j = tid & 255;
    float gi = gate[b * 1024 + j];
    float gf = gate[b * 1024 + 256 + j];
    float gg = gate[b * 1024 + 512 + j];
    float go = gate[b * 1024 + 768 + j];
    float cv = sigf(gf) * c[tid] + sigf(gi) * tanhf(gg);
    float hv = sigf(go) * tanhf(cv);
    __syncthreads();
    h[tid] = hv; c[tid] = cv;
    __syncthreads();
  }
  // base[b] = fc_b + sum_j h[b][j]*fc_w[j]
  gate[tid] = h[tid] * fcw[tid & 255];
  __syncthreads();
  for (int s = 128; s > 0; s >>= 1) {
    if ((tid & 255) < s) gate[tid] += gate[tid + s];
    __syncthreads();
  }
  if ((tid & 255) == 0) base[tid >> 8] = gate[tid] + fcb[0];
}

// ---------------------------------------------------------------------------
// GCN helpers
// ---------------------------------------------------------------------------
__global__ __launch_bounds__(1024) void transpose_zz(const float* __restrict__ zz,
                                                     float* __restrict__ zzT) {
  __shared__ float tile[32][33];
  int nBase = blockIdx.x * 32, btBase = blockIdx.y * 32;
  int tx = threadIdx.x, ty = threadIdx.y;
  int bt = btBase + ty, n = nBase + tx;
  tile[ty][tx] = (bt < BT && n < Nv) ? zz[(size_t)bt * Nv + n] : 0.0f;
  __syncthreads();
  int n2 = nBase + ty, bt2 = btBase + tx;
  if (n2 < Nv && bt2 < BT) zzT[(size_t)n2 * BT + bt2] = tile[tx][ty];
}

__global__ __launch_bounds__(256) void deg_init(float* __restrict__ deg) {
  int i = blockIdx.x * 256 + threadIdx.x;
  if (i < Nv) deg[i] = 1.0f;  // self loop
}

__global__ __launch_bounds__(256) void deg_count(const int* __restrict__ dst,
                                                 float* __restrict__ deg) {
  int e = blockIdx.x * 256 + threadIdx.x;
  if (e < Ev) atomicAdd(&deg[dst[e]], 1.0f);
}

__global__ __launch_bounds__(256) void deg_rsqrt(float* __restrict__ deg) {
  int i = blockIdx.x * 256 + threadIdx.x;
  if (i < Nv) deg[i] = rsqrtf(deg[i]);
}

// self-loop contribution: aggT[n][bt] = zzT[n][bt] * dinv[n]^2
__global__ __launch_bounds__(256) void agg_init(const float* __restrict__ zzT,
                                                const float* __restrict__ dinv,
                                                float* __restrict__ aggT) {
  int i = blockIdx.x * 256 + threadIdx.x;
  if (i >= Nv * BT) return;
  int n = i / BT;
  float dv = dinv[n];
  aggT[i] = zzT[i] * dv * dv;
}

__global__ __launch_bounds__(256) void edge_scatter(const int* __restrict__ src,
                                                    const int* __restrict__ dst,
                                                    const float* __restrict__ zzT,
                                                    const float* __restrict__ dinv,
                                                    float* __restrict__ aggT) {
  int e = blockIdx.x * 256 + threadIdx.x;
  if (e >= Ev) return;
  int s = src[e], d = dst[e];
  float nrm = dinv[s] * dinv[d];
  const float* zp = zzT + (size_t)s * BT;
  float* ap = aggT + (size_t)d * BT;
#pragma unroll 4
  for (int bt = 0; bt < BT; ++bt) atomicAdd(&ap[bt], zp[bt] * nrm);
}

// u[g] = zlstm_wih[g] . gcn_w[0];  v[g] = zlstm_wih[g] . gcn_b + bih[g] + bhh[g]
__global__ __launch_bounds__(512) void uv_kernel(const float* __restrict__ zwih,
                                                 const float* __restrict__ gw,
                                                 const float* __restrict__ gb,
                                                 const float* __restrict__ zbih,
                                                 const float* __restrict__ zbhh,
                                                 float* __restrict__ uv) {
  int g = threadIdx.x;
  float u = 0.0f, v = 0.0f;
  const float* wr = zwih + g * GH;
#pragma unroll 4
  for (int j = 0; j < GH; ++j) { float w = wr[j]; u += w * gw[j]; v += w * gb[j]; }
  uv[g] = u;
  uv[512 + g] = v + zbih[g] + zbhh[g];
}

__global__ __launch_bounds__(256) void whh_to_f16(const float* __restrict__ whh,
                                                  _Float16* __restrict__ whh16) {
  int i = blockIdx.x * 256 + threadIdx.x;
  if (i < 4 * ZLH * ZLH) whh16[i] = (_Float16)whh[i];
}

// ---------------------------------------------------------------------------
// zlstm: 80000 sequences, T=24, in 64 (collapsed to scalar a*u+v), hidden 128.
// Block = 16 sequences, 512 threads = 16 waves; wave owns 32 gate columns.
// Recurrent GEMM H[16x128] @ whh.T[128x512] via v_wmma_f32_16x16x32_f16.
// Fuses final fc dot -> out[b*N+n].
// ---------------------------------------------------------------------------
#define GSTR 516   // gates LDS row stride (floats), bank-conflict pad

__global__ __launch_bounds__(512) void zlstm_kernel(const float* __restrict__ aggT,
                                                    const _Float16* __restrict__ whh16,
                                                    const float* __restrict__ uv,
                                                    const float* __restrict__ fcw,
                                                    const float* __restrict__ base,
                                                    float* __restrict__ out) {
  __shared__ float gatesLds[16 * GSTR];                 // 33.0 KB
  __shared__ float cLds[16 * ZLH];                      // 8 KB
  __shared__ alignas(16) _Float16 hA[16 * 136];         // 4.25 KB, A-layout source
  __shared__ float uLds[512];
  __shared__ float vLds[512];
  __shared__ float aCur[16];

  int tid  = threadIdx.x;
  int wave = tid >> 5, lane = tid & 31;
  int g = lane >> 4, col = lane & 15;
  int s0 = blockIdx.x * 16;
  int b  = s0 / Nv;           // 20000 % 16 == 0 -> no straddle
  int n0 = s0 % Nv;

  // Preload B fragments of whh.T (static across steps): 2 n-tiles x 4 k-tiles.
  v16h Bf0[4], Bf1[4];
#pragma unroll
  for (int kt = 0; kt < 4; ++kt) {
    const _Float16* p0 = whh16 + (size_t)(wave * 32 + col) * ZLH + kt * 32 + 8 * g;
    const _Float16* p1 = whh16 + (size_t)(wave * 32 + 16 + col) * ZLH + kt * 32 + 8 * g;
    Bf0[kt] = pack16(*reinterpret_cast<const v8h*>(p0),
                     *reinterpret_cast<const v8h*>(p0 + 16));
    Bf1[kt] = pack16(*reinterpret_cast<const v8h*>(p1),
                     *reinterpret_cast<const v8h*>(p1 + 16));
  }

  // init h=0, c=0, stage u/v
  for (int i = tid; i < 16 * 136; i += 512) hA[i] = (_Float16)0.0f;
  for (int i = tid; i < 16 * ZLH; i += 512) cLds[i] = 0.0f;
  if (tid < 512) { uLds[tid] = uv[tid]; vLds[tid] = uv[512 + tid]; }
  __syncthreads();

#pragma unroll 1
  for (int t = 0; t < Tv; ++t) {
    if (tid < 16) aCur[tid] = aggT[(size_t)(n0 + tid) * BT + b * Tv + t];
    __syncthreads();

    // A fragments from hA (rows m=col, K = kt*32 + 8g + j / +16)
    v16h Af[4];
#pragma unroll
    for (int kt = 0; kt < 4; ++kt) {
      const _Float16* p = hA + col * 136 + kt * 32 + 8 * g;
      Af[kt] = pack16(*reinterpret_cast<const v8h*>(p),
                      *reinterpret_cast<const v8h*>(p + 16));
    }
    v8f acc0 = {}, acc1 = {};
#pragma unroll
    for (int kt = 0; kt < 4; ++kt) {
      acc0 = wmma_f16(Af[kt], Bf0[kt], acc0);
      acc1 = wmma_f16(Af[kt], Bf1[kt], acc1);
    }

    // gate pre-activations -> LDS: g_pre = wmma + a*u[n] + v[n]
    {
      int n = wave * 32 + col;
      float uu = uLds[n], vv = vLds[n];
      int n2 = n + 16;
      float uu2 = uLds[n2], vv2 = vLds[n2];
#pragma unroll
      for (int r = 0; r < 8; ++r) {
        int m = r + 8 * g;
        float am = aCur[m];
        gatesLds[m * GSTR + n]  = acc0[r] + am * uu  + vv;
        gatesLds[m * GSTR + n2] = acc1[r] + am * uu2 + vv2;
      }
    }
    __syncthreads();

    // LSTM cell update: 16 seq x 128 hidden = 2048 elems over 512 threads
#pragma unroll
    for (int q = 0; q < 4; ++q) {
      int idx = tid + q * 512;
      int s = idx >> 7, j = idx & 127;
      float gi = gatesLds[s * GSTR + j];
      float gf = gatesLds[s * GSTR + 128 + j];
      float gg = gatesLds[s * GSTR + 256 + j];
      float go = gatesLds[s * GSTR + 384 + j];
      float cv = sigf(gf) * cLds[s * ZLH + j] + sigf(gi) * tanhf(gg);
      float hv = sigf(go) * tanhf(cv);
      cLds[s * ZLH + j] = cv;
      hA[s * 136 + j] = (_Float16)hv;
    }
    __syncthreads();
  }

  // out[b, 0, n0+s] = base[b] + h_s . fc_w[256:384]
  if (tid < 16) {
    float acc = base[b];
    const float* f2 = fcw + CLH;
#pragma unroll 4
    for (int j = 0; j < ZLH; ++j) acc += (float)hA[tid * 136 + j] * f2[j];
    out[(size_t)b * Nv + n0 + tid] = acc;
  }
}

// ---------------------------------------------------------------------------
// Host launcher
// ---------------------------------------------------------------------------
static constexpr size_t al256(size_t x) { return (x + 255) & ~(size_t)255; }

extern "C" void kernel_launch(void* const* d_in, const int* in_sizes, int n_in,
                              void* d_out, int out_size, void* d_ws, size_t ws_size,
                              hipStream_t stream) {
  (void)in_sizes; (void)n_in; (void)out_size; (void)ws_size;

  const float* era5 = (const float*)d_in[0];
  const float* zeta = (const float*)d_in[1];
  const int*   eidx = (const int*)d_in[2];
  const float* c1w  = (const float*)d_in[3];
  const float* c1b  = (const float*)d_in[4];
  const float* c2w  = (const float*)d_in[5];
  const float* c2b  = (const float*)d_in[6];
  const float* cwih = (const float*)d_in[7];
  const float* cwhh = (const float*)d_in[8];
  const float* cbih = (const float*)d_in[9];
  const float* cbhh = (const float*)d_in[10];
  const float* gw   = (const float*)d_in[11];
  const float* gb   = (const float*)d_in[12];
  const float* zwih = (const float*)d_in[13];
  const float* zwhh = (const float*)d_in[14];
  const float* zbih = (const float*)d_in[15];
  const float* zbhh = (const float*)d_in[16];
  const float* fcw  = (const float*)d_in[17];
  const float* fcb  = (const float*)d_in[18];
  float* out = (float*)d_out;

  const int* srcp = eidx;
  const int* dstp = eidx + Ev;

  char* ws = (char*)d_ws;
  size_t off = 0;
  float*    convA = (float*)(ws + off);   off = al256(off + (size_t)CONVA_ELEMS * 4);
  _Float16* fA    = (_Float16*)(ws + off); off = al256(off + (size_t)BT * FLAT * 2);
  float*    xW    = (float*)(ws + off);   off = al256(off + (size_t)BT * 1024 * 4);
  float*    base  = (float*)(ws + off);   off = al256(off + 256);
  float*    dinv  = (float*)(ws + off);   off = al256(off + (size_t)Nv * 4);
  float*    zzT   = (float*)(ws + off);   off = al256(off + (size_t)Nv * BT * 4);
  float*    aggT  = (float*)(ws + off);   off = al256(off + (size_t)Nv * BT * 4);
  float*    uv    = (float*)(ws + off);   off = al256(off + 1024 * 4);
  _Float16* whh16 = (_Float16*)(ws + off); off = al256(off + (size_t)4 * ZLH * ZLH * 2);

  // CNN branch
  conv1_pool<<<(CONVA_ELEMS + 255) / 256, 256, 0, stream>>>(era5, c1w, c1b, convA);
  conv2_pool<<<(BT * CH2 * P2H * P2W + 255) / 256, 256, 0, stream>>>(convA, c2w, c2b, fA);
  xw_wmma<<<48, 256, 0, stream>>>(fA, cwih, cbih, cbhh, xW);
  clstm_kernel<<<1, 1024, 0, stream>>>(xW, cwhh, fcw, fcb, base);

  // GCN branch
  transpose_zz<<<dim3(Nv / 32, BT / 32), dim3(32, 32), 0, stream>>>(zeta, zzT);
  deg_init<<<(Nv + 255) / 256, 256, 0, stream>>>(dinv);
  deg_count<<<Ev / 256, 256, 0, stream>>>(dstp, dinv);
  deg_rsqrt<<<(Nv + 255) / 256, 256, 0, stream>>>(dinv);
  agg_init<<<(Nv * BT) / 256, 256, 0, stream>>>(zzT, dinv, aggT);
  edge_scatter<<<Ev / 256, 256, 0, stream>>>(srcp, dstp, zzT, dinv, aggT);

  // zlstm prep + main WMMA kernel (fuses final fc -> out)
  uv_kernel<<<1, 512, 0, stream>>>(zwih, gw, gb, zbih, zbhh, uv);
  whh_to_f16<<<(4 * ZLH * ZLH) / 256, 256, 0, stream>>>(zwhh, whh16);
  zlstm_kernel<<<(Bv * Nv) / 16, 512, 0, stream>>>(aggT, whh16, uv, fcw, base, out);
}
